// cheb_conv_withSAt_32512902431509
// MI455X (gfx1250) — matrix operand order repaired and checked by reading, not compiled, not run on hardware
//
#include <hip/hip_runtime.h>

// ---------------------------------------------------------------------------
// cheb_conv_withSAt on MI455X (gfx1250, wave32, WMMA)
//
// out[b,n,(o,t)] = relu( sum_{k,i} (cheb[k,i,n]*attn[b,i,n]) * xth[b,k,i,(o,t)] )
// Batched GEMM: M=2048(n) x N=384(o,t) x K=6144(k,i) per batch, 77.3 GFLOP over
// ~235 MB unique bytes (~330 FLOP/B) -> matrix-core bound on MI455X.
// Precision: bf16 hi/lo split of both operands, f32 accumulate
// (3x v_wmma_f32_16x16x32_bf16 per fp32 tile, ~2^-16 relative error).
// cheb (50MB) + attn slice (16MB) + packed xth (75MB) all fit in 192MB L2.
// ---------------------------------------------------------------------------

#define B_      8
#define N_      2048
#define FIN     32
#define FOUT    32
#define T_      12
#define K_      3
#define OT      (FOUT * T_)      // 384 fused (o,t) columns
#define ICHUNKS (N_ / 32)        // 64 chunks of 32 along the i contraction
#define OTTILES (OT / 16)        // 24 wmma column tiles
#define FRAG    512              // 32(i) x 16(ot) bf16 elements per fragment blk

typedef __attribute__((ext_vector_type(16))) __bf16 v16bf;
typedef __attribute__((ext_vector_type(8)))  __bf16 v8bf;
typedef __attribute__((ext_vector_type(8)))  float  v8f;

// ---------------------------------------------------------------------------
// Kernel 1: xth[b,k,i,(o,t)] = sum_f x[b,i,f,t] * theta[k,f,o], written split
// (hi/lo bf16) directly in the WMMA B-matrix lane layout:
//   B 32x16 bf16 fragment: lane = 16*h + col, element e -> i = ic*32 + 16h + e
//   flat = ((((b*3+k)*64 + ic)*24 + otc)*512 + lane*16 + e
// One block per (b, ic): x slice (48KB) + theta (12KB) staged in LDS once, so
// x is read from HBM exactly once; every thread emits contiguous 32B stores.
// ---------------------------------------------------------------------------
#define XTH_THREADS 384
__global__ __launch_bounds__(XTH_THREADS)
void xth_pack_kernel(const float* __restrict__ x,
                     const float* __restrict__ theta,
                     unsigned short* __restrict__ wsh,
                     unsigned short* __restrict__ wsl) {
    __bf16* ws_hi = (__bf16*)wsh;
    __bf16* ws_lo = (__bf16*)wsl;
    const int ic  = blockIdx.x;   // 64
    const int b   = blockIdx.y;   // 8
    const int tid = threadIdx.x;  // 384 threads = one (o,t) column each

    __shared__ float xs[384 * 33];          // [f*T+t][ii], pad 33: conflict-free
    __shared__ float th[K_ * FIN * FOUT];   // theta, 12KB

    // stage x[b, ic*32 .. +31, :, :]: rows of 384 contiguous floats, transposed
    const float* xg = x + ((long)(b * N_) + (long)ic * 32) * (FIN * T_);
    for (int s = tid; s < 32 * 384; s += XTH_THREADS) {
        int ii = s / 384;
        int ft = s % 384;
        xs[ft * 33 + ii] = xg[(long)ii * 384 + ft];
    }
    for (int s = tid; s < K_ * FIN * FOUT; s += XTH_THREADS)
        th[s] = theta[s];
    __syncthreads();

    const int ot  = tid;
    const int o   = ot / T_;
    const int t   = ot % T_;
    const int otc = ot >> 4;
    const int col = ot & 15;

    for (int k = 0; k < K_; ++k) {
#pragma unroll
        for (int h = 0; h < 2; ++h) {       // i half within 32-chunk
            float acc[16];
#pragma unroll
            for (int e = 0; e < 16; ++e) acc[e] = 0.f;
            for (int f = 0; f < FIN; ++f) {
                float tv = th[(k * FIN + f) * FOUT + o];
                const float* xr = &xs[(f * T_ + t) * 33 + h * 16];
#pragma unroll
                for (int e = 0; e < 16; ++e)
                    acc[e] += xr[e] * tv;
            }
            long base = ((((long)(b * K_ + k)) * ICHUNKS + ic) * OTTILES + otc)
                        * (long)FRAG + (h * 16 + col) * 16;
            v16bf vh, vl;
#pragma unroll
            for (int e = 0; e < 16; ++e) {
                __bf16 hi = (__bf16)acc[e];
                vh[e] = hi;
                vl[e] = (__bf16)(acc[e] - (float)hi);
            }
            *(v16bf*)(ws_hi + base) = vh;   // 32B contiguous per thread
            *(v16bf*)(ws_lo + base) = vl;
        }
    }
}

// ---------------------------------------------------------------------------
// Kernel 2: the graph contraction. Block: 256 threads = 8 waves.
// Block tile: 32 n rows x 384 ot cols; wave w owns ot tiles [3w,3w+2] and BOTH
// 16-row n subtiles, so each B fragment load feeds 2 A subtiles -> 18 WMMAs
// per 32-wide i chunk against 6 B-fragment loads (halves L2 B traffic).
// A tile (cheb*attn fused, hi/lo split) built in LDS, transposed [n][i] with
// row stride 40 bf16 -> each lane's fragment halves are aligned ds_load_b128,
// bank-conflict-free (bank stride 20).
// ---------------------------------------------------------------------------
__global__ __launch_bounds__(256)
void cheb_sat_wmma_kernel(const float* __restrict__ attn,
                          const float* __restrict__ cheb,
                          const unsigned short* __restrict__ wsh,
                          const unsigned short* __restrict__ wsl,
                          float* __restrict__ out) {
    const __bf16* xh = (const __bf16*)wsh;
    const __bf16* xl = (const __bf16*)wsl;

    const int n0   = blockIdx.x * 32;
    const int b    = blockIdx.y;
    const int tid  = threadIdx.x;
    const int wave = tid >> 5;
    const int lane = tid & 31;
    const int h    = lane >> 4;   // half-wave select
    const int m    = lane & 15;   // row within 16 (A/D), col within 16 (B)

    __shared__ __align__(16) __bf16 aHi[32 * 40];
    __shared__ __align__(16) __bf16 aLo[32 * 40];

    v8f acc[2][3] = {};
    const long N2 = (long)N_ * N_;

    for (int k = 0; k < K_; ++k) {
        const float* chebk = cheb + (long)k * N2;
        const float* attnb = attn + (long)b * N2;
        for (int ic = 0; ic < ICHUNKS; ++ic) {
            // --- cooperative fill: 32(i) x 32(n) fused weights, float2 loads ---
#pragma unroll
            for (int s = tid; s < 512; s += 256) {
                int ii = s >> 4;            // i within chunk
                int nn = (s & 15) * 2;      // n pair within tile
                long g = (long)(ic * 32 + ii) * N_ + (n0 + nn);
                float2 c2 = *(const float2*)(chebk + g);
                float2 a2 = *(const float2*)(attnb + g);
                float w0 = c2.x * a2.x, w1 = c2.y * a2.y;
                __bf16 h0 = (__bf16)w0, h1 = (__bf16)w1;
                aHi[nn * 40 + ii]       = h0;
                aLo[nn * 40 + ii]       = (__bf16)(w0 - (float)h0);
                aHi[(nn + 1) * 40 + ii] = h1;
                aLo[(nn + 1) * 40 + ii] = (__bf16)(w1 - (float)h1);
            }
            if (ic + 1 < ICHUNKS) {   // uniform branch: warm L2 for next chunk
                long gp = (long)(ic * 32 + 32 + (tid >> 3)) * N_ + n0 + (tid & 7) * 4;
                __builtin_prefetch(chebk + gp, 0, 1);
                __builtin_prefetch(attnb + gp, 0, 1);
            }
            __syncthreads();

            // --- A fragments (16x32 bf16) for both n subtiles:
            //     lane m,h; element e -> K=16h+e ---
            v16bf ah[2], al[2];
#pragma unroll
            for (int u = 0; u < 2; ++u) {
                const int row = (u * 16 + m) * 40;
                v8bf r0 = *(const v8bf*)&aHi[row + 16 * h];
                v8bf r1 = *(const v8bf*)&aHi[row + 16 * h + 8];
                v8bf s0 = *(const v8bf*)&aLo[row + 16 * h];
                v8bf s1 = *(const v8bf*)&aLo[row + 16 * h + 8];
#pragma unroll
                for (int e = 0; e < 8; ++e) {
                    ah[u][e] = r0[e]; ah[u][e + 8] = r1[e];
                    al[u][e] = s0[e]; al[u][e + 8] = s1[e];
                }
            }

            // --- B fragments (pre-packed, one aligned 32B load per lane),
            //     each feeds both n subtiles: 18 WMMAs / chunk ---
            long base = ((((long)(b * K_ + k)) * ICHUNKS + ic) * OTTILES + wave * 3)
                        * (long)FRAG + lane * 16;
#pragma unroll
            for (int j = 0; j < 3; ++j) {
                const v16bf bh = *(const v16bf*)(xh + base + j * FRAG);
                const v16bf bl = *(const v16bf*)(xl + base + j * FRAG);
#pragma unroll
                for (int u = 0; u < 2; ++u) {
                    acc[u][j] = __builtin_amdgcn_wmma_f32_16x16x32_bf16(
                        false, ah[u], false, bh, (short)0, acc[u][j], false, false);
                    acc[u][j] = __builtin_amdgcn_wmma_f32_16x16x32_bf16(
                        false, ah[u], false, bl, (short)0, acc[u][j], false, false);
                    acc[u][j] = __builtin_amdgcn_wmma_f32_16x16x32_bf16(
                        false, al[u], false, bh, (short)0, acc[u][j], false, false);
                }
            }
            __syncthreads();
        }
    }

    // --- epilogue: relu + store. D layout: VGPR r -> row r + 8h, col m. ---
#pragma unroll
    for (int u = 0; u < 2; ++u) {
#pragma unroll
        for (int j = 0; j < 3; ++j) {
            const int otc = wave * 3 + j;
#pragma unroll
            for (int r = 0; r < 8; ++r) {
                int n = n0 + u * 16 + r + 8 * h;
                long off = (long)(b * N_ + n) * OT + otc * 16 + m;
                out[off] = fmaxf(acc[u][j][r], 0.0f);
            }
        }
    }
}

// ---------------------------------------------------------------------------
// inputs (setup_inputs order): x [8,2048,32,12] f32, spatial_attention
// [8,2048,2048] f32, cheb [3,2048,2048] f32, theta [3,32,32] f32.
// output: [8,2048,32,12] f32.  ws: 2 planes * 18,874,368 bf16 = 75.5 MB.
// ---------------------------------------------------------------------------
extern "C" void kernel_launch(void* const* d_in, const int* in_sizes, int n_in,
                              void* d_out, int out_size, void* d_ws, size_t ws_size,
                              hipStream_t stream) {
    const float* x     = (const float*)d_in[0];
    const float* attn  = (const float*)d_in[1];
    const float* cheb  = (const float*)d_in[2];
    const float* theta = (const float*)d_in[3];
    float* out = (float*)d_out;

    const long plane = (long)B_ * K_ * ICHUNKS * OTTILES * FRAG;  // 18,874,368
    unsigned short* ws_hi = (unsigned short*)d_ws;
    unsigned short* ws_lo = ws_hi + plane;

    xth_pack_kernel<<<dim3(ICHUNKS, B_), dim3(XTH_THREADS), 0, stream>>>(
        x, theta, ws_hi, ws_lo);

    cheb_sat_wmma_kernel<<<dim3(N_ / 32, B_), dim3(256), 0, stream>>>(
        attn, cheb, ws_hi, ws_lo, out);
}
